// Latent_83751862272678
// MI455X (gfx1250) — compile-verified, matrix operand order checked by reading, not run on hardware
//
#include <hip/hip_runtime.h>
#include <hip/hip_bf16.h>
#include <stdint.h>

// Problem constants (from reference: z [4,64,64,64], e [4096,64])
#define N_PTS   16384
#define M_CODES 4096
#define D_DIM   64
#define NSPLIT  8      // split of N across grid.y
#define MTILE   64     // columns per workgroup
#define NTILE   128    // rows per workgroup iteration (8 waves x 16 rows)

#define LOG2E   1.44269504088896340736f

typedef __attribute__((ext_vector_type(16))) __bf16 v16bf;
typedef __attribute__((ext_vector_type(8)))  float  v8f;

static __device__ __forceinline__ uint16_t f2bf(float f) {
  uint32_t u = __float_as_uint(f);
  u += 0x7FFFu + ((u >> 16) & 1u);   // round-to-nearest-even
  return (uint16_t)(u >> 16);
}

union FragU { uint4 q[2]; v16bf v; };

// Per-lane bf16 fragment for a K=32 chunk starting at k0 of a row-major row.
// ISA 7.12.2 (16-bit A 16x32): lane hi-half holds K in [hi*8, hi*8+8) U [16+hi*8, ...).
static __device__ __forceinline__ v16bf load_frag(const uint16_t* row, int hi, int k0) {
  FragU u;
  u.q[0] = *reinterpret_cast<const uint4*>(row + k0 + hi * 8);
  u.q[1] = *reinterpret_cast<const uint4*>(row + k0 + 16 + hi * 8);
  return u.v;
}

// ---- prep codebook: bf16 copy + ||e||^2 ----
__global__ void lat_prep_e(const float* __restrict__ e,
                           uint16_t* __restrict__ ebf,
                           float* __restrict__ em2) {
  __shared__ float red[64];
  const int m = blockIdx.x, d = threadIdx.x;
  const float v = e[m * D_DIM + d];
  ebf[m * D_DIM + d] = f2bf(v);
  red[d] = v * v;
  __syncthreads();
  for (int off = 32; off > 0; off >>= 1) {
    if (d < off) red[d] += red[d + off];
    __syncthreads();
  }
  if (d == 0) em2[m] = red[0];
}

// ---- prep latents: [B,D,H,W] -> row-major bf16 [N,D] + ||z||^2 (LDS-tiled transpose) ----
__global__ void lat_prep_z(const float* __restrict__ z,
                           uint16_t* __restrict__ zbf,
                           float* __restrict__ zn2) {
  __shared__ float tile[64][65];
  const int t = threadIdx.x;
  const int b = blockIdx.x >> 6;
  const int hw0 = (blockIdx.x & 63) << 6;
  const float* zb = z + b * (D_DIM * 4096) + hw0;
#pragma unroll
  for (int k = 0; k < 16; ++k) {               // coalesced reads along hw
    const int idx = k * 256 + t;
    const int d = idx >> 6, noff = idx & 63;
    tile[noff][d] = zb[d * 4096 + noff];
  }
  __syncthreads();
  const int row = t >> 2;
  const int j0 = (t & 3) * 16;
  const int n = b * 4096 + hw0 + row;
  float ss = 0.f;
#pragma unroll
  for (int j = 0; j < 16; ++j) {               // coalesced-ish writes along d
    const float v = tile[row][j0 + j];
    ss += v * v;
    zbf[n * D_DIM + j0 + j] = f2bf(v);
  }
  ss += __shfl_xor(ss, 1);                     // 4 lanes share one row
  ss += __shfl_xor(ss, 2);
  if ((t & 3) == 0) zn2[n] = ss;
}

// ---- fused GEMM + exp + deterministic column reduction ----
// Per column m: S_m = sum_n exp(alpha*(zn2[n] + em2[m] - 2*dot(z_n,e_m)))
// All exponents <= 0 (alpha<0, d2>=0) so a streaming sum needs no max pass.
// Exponent is produced directly in base-2: x2 = aL*zn2 + aL*em2 + tL*dot,
// with aL = alpha*log2e, tL = -2*alpha*log2e, consumed by raw v_exp_f32 (2^x).
__global__ void __launch_bounds__(256)
lat_gemm_lse(const uint16_t* __restrict__ zbf,
             const uint16_t* __restrict__ ebf,
             const float* __restrict__ zn2,
             const float* __restrict__ em2,
             const float* __restrict__ log_sigma,
             float* __restrict__ Spart) {
  __shared__ float colred[8][64];
  const float ls = log_sigma[0];
  const float t2 = __expf(-2.f * ls);          // -2*alpha = exp(-2*ls)
  const float tL = t2 * LOG2E;                 //  (-2*alpha)*log2e
  const float aL = -0.5f * tL;                 //  alpha*log2e
  const int w  = threadIdx.x >> 5;             // wave id (8 waves)
  const int l  = threadIdx.x & 31;
  const int hi = l >> 4, lo = l & 15;
  const int m0 = blockIdx.x * MTILE;
  const int nstart = blockIdx.y * (N_PTS / NSPLIT);

  float aem2c[4];                              // aL*||e||^2, lane column constants
#pragma unroll
  for (int c = 0; c < 4; ++c) aem2c[c] = aL * em2[m0 + c * 16 + lo];

  float acc[4] = {0.f, 0.f, 0.f, 0.f};         // streaming sum of exp per column

  for (int it = 0; it < (N_PTS / NSPLIT) / NTILE; ++it) {
    const int nbase = nstart + it * NTILE + w * 16;   // this wave's 16-row stripe
    const uint16_t* arow = zbf + (nbase + lo) * D_DIM;
    const v16bf a0 = load_frag(arow, hi, 0);
    const v16bf a1 = load_frag(arow, hi, 32);
    // ||z||^2 for this lane's 8 C rows (row = nbase + r + 8*hi)
    const float4 zq0 = *reinterpret_cast<const float4*>(zn2 + nbase + hi * 8);
    const float4 zq1 = *reinterpret_cast<const float4*>(zn2 + nbase + hi * 8 + 4);
    const float zr[8] = {zq0.x, zq0.y, zq0.z, zq0.w, zq1.x, zq1.y, zq1.z, zq1.w};
#pragma unroll
    for (int c = 0; c < 4; ++c) {
      const uint16_t* brow = ebf + (m0 + c * 16 + lo) * D_DIM;  // B = e^T, col-major == e row-major
      const v16bf b0 = load_frag(brow, hi, 0);
      const v16bf b1 = load_frag(brow, hi, 32);
      v8f cf = {};
      cf = __builtin_amdgcn_wmma_f32_16x16x32_bf16(false, a0, false, b0, (short)0, cf, false, false);
      cf = __builtin_amdgcn_wmma_f32_16x16x32_bf16(false, a1, false, b1, (short)0, cf, false, false);
      float s = 0.f;
#pragma unroll
      for (int r = 0; r < 8; ++r) {
        // 2 FMAs + 1 v_exp_f32 per matrix element (minimum for this epilogue)
        const float x2 = __builtin_fmaf(tL, cf[r], __builtin_fmaf(aL, zr[r], aem2c[c]));
        s += __builtin_amdgcn_exp2f(x2);
      }
      acc[c] += s;
    }
  }
  // fold lane pairs (row halves) then combine waves in a fixed order (deterministic)
#pragma unroll
  for (int c = 0; c < 4; ++c) {
    const float p = acc[c] + __shfl_xor(acc[c], 16);
    if (hi == 0) colred[w][c * 16 + lo] = p;
  }
  __syncthreads();
  if (threadIdx.x < 64) {
    float s = 0.f;
#pragma unroll
    for (int ww = 0; ww < 8; ++ww) s += colred[ww][threadIdx.x];
    Spart[blockIdx.y * M_CODES + m0 + threadIdx.x] = s;
  }
}

// ---- finalize: loss = -mean(log S_m) + 0.5*D*(2*ls - 1) + log N ----
__global__ void lat_finalize(const float* __restrict__ Spart,
                             const float* __restrict__ log_sigma,
                             float* __restrict__ out) {
  __shared__ float red[256];
  const int t = threadIdx.x;
  float s = 0.f;
  for (int m = t; m < M_CODES; m += 256) {
    float S = 0.f;
#pragma unroll
    for (int ny = 0; ny < NSPLIT; ++ny) S += Spart[ny * M_CODES + m];
    s += __logf(S);
  }
  red[t] = s;
  __syncthreads();
  for (int off = 128; off > 0; off >>= 1) {
    if (t < off) red[t] += red[t + off];
    __syncthreads();
  }
  if (t == 0) {
    const float ls = log_sigma[0];
    out[0] = -red[0] / (float)M_CODES
             + 0.5f * (float)D_DIM * (2.f * ls - 1.f)
             + __logf((float)N_PTS);
  }
}

extern "C" void kernel_launch(void* const* d_in, const int* in_sizes, int n_in,
                              void* d_out, int out_size, void* d_ws, size_t ws_size,
                              hipStream_t stream) {
  const float* z  = (const float*)d_in[0];
  const float* e  = (const float*)d_in[1];
  const float* ls = (const float*)d_in[2];
  char* ws = (char*)d_ws;
  uint16_t* zbf   = (uint16_t*)(ws);             // 16384*64*2 = 2 MiB
  uint16_t* ebf   = (uint16_t*)(ws + 2097152);   // 4096*64*2  = 512 KiB
  float*    zn2   = (float*)(ws + 2621440);      // 64 KiB
  float*    em2   = (float*)(ws + 2686976);      // 16 KiB
  float*    Spart = (float*)(ws + 2703360);      // 8*4096*4 = 128 KiB

  lat_prep_e<<<M_CODES, 64, 0, stream>>>(e, ebf, em2);
  lat_prep_z<<<256, 256, 0, stream>>>(z, zbf, zn2);
  lat_gemm_lse<<<dim3(M_CODES / MTILE, NSPLIT), 256, 0, stream>>>(zbf, ebf, zn2, em2, ls, Spart);
  lat_finalize<<<1, 256, 0, stream>>>(Spart, ls, (float*)d_out);
}
